// MBDiscriminator_13606456394356
// MI455X (gfx1250) — matrix operand order repaired and checked by reading, not compiled, not run on hardware
//
#include <hip/hip_runtime.h>
#include <hip/hip_bf16.h>

// MI455X (gfx1250) implementation.
// conv2 + minibatch GEMM on v_wmma_f32_16x16x32_f16 (f16 in, f32 accumulate).
// The minibatch GEMM register-blocks the whole M=64 batch dimension per wave
// (4x4 fragment grid) so T (the dominant byte stream) is read exactly once.

typedef __attribute__((ext_vector_type(16))) _Float16 v16h;
typedef __attribute__((ext_vector_type(8)))  _Float16 v8h;
typedef __attribute__((ext_vector_type(8)))  float    v8f;

union FragAB { v16h v; v8h h[2]; _Float16 e[16]; };
union FragC  { v8f  v; float e[8]; };

#define FLAT   131072   // 128 * 32 * 32
#define BATCH  64

// ---------------- conv1: 3->64, 4x4 s2 p1, leaky(0.2), emit f16 -------------
__global__ void k_conv1(const float* __restrict__ x, const float* __restrict__ w,
                        const float* __restrict__ bias, _Float16* __restrict__ h1) {
  int blk = blockIdx.x;              // 64 b * 64 oc * 16 chunks
  int schunk = blk & 15;
  int oc = (blk >> 4) & 63;
  int b  = blk >> 10;
  int s  = schunk * 256 + threadIdx.x;     // 0..4095
  int oy = s >> 6, ox = s & 63;
  float acc = bias[oc];
  const float* wp = w + oc * 48;           // [3][4][4]
  #pragma unroll
  for (int ci = 0; ci < 3; ++ci) {
    const float* xp = x + (size_t)(b * 3 + ci) * 128 * 128;
    #pragma unroll
    for (int ky = 0; ky < 4; ++ky) {
      int iy = 2 * oy - 1 + ky;
      if ((unsigned)iy >= 128u) continue;
      #pragma unroll
      for (int kx = 0; kx < 4; ++kx) {
        int ix = 2 * ox - 1 + kx;
        if ((unsigned)ix >= 128u) continue;
        acc += xp[iy * 128 + ix] * wp[ci * 16 + ky * 4 + kx];
      }
    }
  }
  acc = acc >= 0.f ? acc : 0.2f * acc;
  h1[(size_t)(b * 64 + oc) * 4096 + s] = (_Float16)acc;
}

// ---------------- convert conv2 weights [128][1024] f32 -> f16 --------------
__global__ void k_cvtW(const float* __restrict__ w, _Float16* __restrict__ wh) {
  int i = blockIdx.x * 256 + threadIdx.x;      // 131072
  wh[i] = (_Float16)w[i];
}

// ---------------- conv2 as implicit-im2col WMMA GEMM ------------------------
// M=128 oc, N=16 pixels per block, K=1024. Block: 256 thr = 8 waves (1 M-tile
// each). B tile (32x16 per K-chunk) staged in LDS as [n][k]; padding handled
// branchlessly (clamped load + select) to keep the staging straight-line.
__global__ void k_conv2(const _Float16* __restrict__ h1, const _Float16* __restrict__ wA,
                        const float* __restrict__ bias, float* __restrict__ h2) {
  __shared__ __align__(16) _Float16 Bl[16][40];   // stride 40 halfs = 80B
  int b  = blockIdx.x >> 6;
  int s0 = (blockIdx.x & 63) * 16;
  int lane = threadIdx.x & 31;
  int wave = threadIdx.x >> 5;                    // M-tile id 0..7
  int l15 = lane & 15;
  int hi  = lane >> 4;

  FragC acc;
  #pragma unroll
  for (int i = 0; i < 8; ++i) acc.e[i] = bias[wave * 16 + i + hi * 8];

  // this thread stages im2col elements (n, kr) and (n, kr+16)
  int n   = threadIdx.x & 15;
  int kr  = threadIdx.x >> 4;                     // 0..15
  int pix = s0 + n;
  int oy = pix >> 5, ox = pix & 31;

  for (int kc = 0; kc < 32; ++kc) {
    int K32 = kc * 32;
    __syncthreads();
    #pragma unroll
    for (int e = 0; e < 2; ++e) {
      int k  = kr + e * 16;                       // 0..31 in chunk
      int Kg = K32 + k;
      int c  = Kg >> 4;
      int r  = Kg & 15;
      int ky = r >> 2, kx = r & 3;
      int iy = 2 * oy - 1 + ky;
      int ix = 2 * ox - 1 + kx;
      bool inb = ((unsigned)iy < 64u) && ((unsigned)ix < 64u);
      int iyc = min(max(iy, 0), 63);
      int ixc = min(max(ix, 0), 63);
      _Float16 val = h1[(size_t)(b * 64 + c) * 4096 + iyc * 64 + ixc];
      Bl[n][k] = inb ? val : (_Float16)0.f;
    }
    __syncthreads();
    FragAB a, bf;
    const _Float16* ap = wA + (size_t)(wave * 16 + l15) * 1024 + K32 + hi * 8;
    a.h[0] = *(const v8h*)(ap);
    a.h[1] = *(const v8h*)(ap + 16);
    const _Float16* bp = &Bl[l15][hi * 8];
    bf.h[0] = *(const v8h*)(bp);
    bf.h[1] = *(const v8h*)(bp + 16);
    acc.v = __builtin_amdgcn_wmma_f32_16x16x32_f16(false, a.v, false, bf.v,
                                                   (short)0, acc.v, false, false);
  }
  #pragma unroll
  for (int i = 0; i < 8; ++i) {
    int oc = wave * 16 + i + hi * 8;
    h2[(size_t)b * FLAT + oc * 1024 + (s0 + l15)] = acc.e[i];
  }
}

// ---------------- batchnorm stats: one block per channel --------------------
__global__ void k_bnstats(const float* __restrict__ h2, float* __restrict__ mu,
                          float* __restrict__ rstd) {
  __shared__ float s1[256], s2[256];
  int c = blockIdx.x;
  float sum = 0.f, sq = 0.f;
  for (int i = threadIdx.x; i < BATCH * 1024; i += 256) {
    int bi = i >> 10, s = i & 1023;
    float v = h2[(size_t)bi * FLAT + c * 1024 + s];
    sum += v; sq += v * v;
  }
  s1[threadIdx.x] = sum; s2[threadIdx.x] = sq;
  __syncthreads();
  for (int off = 128; off > 0; off >>= 1) {
    if (threadIdx.x < off) { s1[threadIdx.x] += s1[threadIdx.x + off];
                             s2[threadIdx.x] += s2[threadIdx.x + off]; }
    __syncthreads();
  }
  if (threadIdx.x == 0) {
    float m   = s1[0] * (1.f / 65536.f);
    float var = s2[0] * (1.f / 65536.f) - m * m;
    mu[c]   = m;
    rstd[c] = rsqrtf(var + 1e-5f);
  }
}

// ---------------- BN apply + leaky, emit f32 (in place) + f16 ---------------
__global__ void k_bnapply(float* __restrict__ h2, _Float16* __restrict__ f16,
                          const float* __restrict__ mu, const float* __restrict__ rstd,
                          const float* __restrict__ gamma, const float* __restrict__ beta) {
  int idx = blockIdx.x * 256 + threadIdx.x;     // 8388608 total
  int c = (idx >> 10) & 127;
  float v = h2[idx];
  v = (v - mu[c]) * rstd[c] * gamma[c] + beta[c];
  v = v >= 0.f ? v : 0.2f * v;
  h2[idx]  = v;
  f16[idx] = (_Float16)v;
}

// ---------------- transpose-convert T [K=131072][500] -> f16 [512][K] -------
__global__ void k_cvtT(const float* __restrict__ T, _Float16* __restrict__ Tt) {
  __shared__ float tile[32][33];
  int k0 = blockIdx.x * 32;          // 4096 tiles
  int n0 = blockIdx.y * 32;          // 16 tiles (512 padded cols)
  for (int r = threadIdx.y; r < 32; r += 8) {
    int k = k0 + r, nn = n0 + threadIdx.x;
    tile[r][threadIdx.x] = (nn < 500) ? T[(size_t)k * 500 + nn] : 0.f;
  }
  __syncthreads();
  for (int r = threadIdx.y; r < 32; r += 8) {
    int nn = n0 + r;
    int k  = k0 + threadIdx.x;
    Tt[(size_t)nn * FLAT + k] = (_Float16)tile[threadIdx.x][r];
  }
}

__global__ void k_zero(float* __restrict__ p, int nel) {
  int i = blockIdx.x * 256 + threadIdx.x;
  if (i < nel) p[i] = 0.f;
}

// ---------------- M = feat @ T2 : WMMA, K split across blocks ----------------
// One wave per block register-blocks the FULL M=64 batch dim (4 M-frags) and
// a 64-wide N supertile (4 N-frags) -> 16 accumulators, 16 wmma per K-chunk.
// Tt is read exactly once across the grid; feat16 once per N-supertile.
// Grid: x = 8 N-supertiles (512 padded cols), y = 64 K-slices (64 chunks each).
__global__ void k_mbd_gemm(const _Float16* __restrict__ feat16,
                           const _Float16* __restrict__ Tt,
                           float* __restrict__ Mbuf) {
  int nt4 = blockIdx.x;               // 0..7
  int ks  = blockIdx.y;               // 0..63
  int lane = threadIdx.x & 31;
  int l15 = lane & 15, hi = lane >> 4;

  FragC acc[4][4];
  #pragma unroll
  for (int mf = 0; mf < 4; ++mf)
    #pragma unroll
    for (int nf = 0; nf < 4; ++nf)
      #pragma unroll
      for (int i = 0; i < 8; ++i) acc[mf][nf].e[i] = 0.f;

  int K0 = ks * 64 * 32;              // 64 chunks of 32 per slice
  for (int kc = 0; kc < 64; ++kc) {
    int K32 = K0 + kc * 32;
    FragAB a[4], bf[4];
    #pragma unroll
    for (int mf = 0; mf < 4; ++mf) {
      const _Float16* ap = feat16 + (size_t)(mf * 16 + l15) * FLAT + K32 + hi * 8;
      a[mf].h[0] = *(const v8h*)(ap);
      a[mf].h[1] = *(const v8h*)(ap + 16);
      __builtin_prefetch(ap + 32, 0, 1);
    }
    #pragma unroll
    for (int nf = 0; nf < 4; ++nf) {
      const _Float16* bp = Tt + (size_t)(nt4 * 64 + nf * 16 + l15) * FLAT + K32 + hi * 8;
      bf[nf].h[0] = *(const v8h*)(bp);
      bf[nf].h[1] = *(const v8h*)(bp + 16);
      __builtin_prefetch(bp + 32, 0, 1);
    }
    #pragma unroll
    for (int mf = 0; mf < 4; ++mf)
      #pragma unroll
      for (int nf = 0; nf < 4; ++nf)
        acc[mf][nf].v = __builtin_amdgcn_wmma_f32_16x16x32_f16(
            false, a[mf].v, false, bf[nf].v, (short)0, acc[mf][nf].v, false, false);
  }
  // K-split accumulation via f32 global atomics (rolled loops: small code)
  for (int mf = 0; mf < 4; ++mf)
    for (int nf = 0; nf < 4; ++nf)
      for (int i = 0; i < 8; ++i) {
        int m  = mf * 16 + i + hi * 8;
        int nn = nt4 * 64 + nf * 16 + l15;
        __hip_atomic_fetch_add(&Mbuf[m * 512 + nn], acc[mf][nf].e[i],
                               __ATOMIC_RELAXED, __HIP_MEMORY_SCOPE_AGENT);
      }
}

// ---------------- minibatch discrimination: o_b[b,out] ----------------------
__global__ void k_ob(const float* __restrict__ Mbuf, float* __restrict__ ob) {
  int idx = blockIdx.x * 64 + threadIdx.x;     // 6400
  if (idx >= 6400) return;
  int out = idx % 100, bb = idx / 100;
  float m[5];
  #pragma unroll
  for (int kd = 0; kd < 5; ++kd) m[kd] = Mbuf[bb * 512 + out * 5 + kd];
  float acc = 0.f;
  for (int a = 0; a < 64; ++a) {
    float norm = 0.f;
    #pragma unroll
    for (int kd = 0; kd < 5; ++kd) norm += fabsf(m[kd] - Mbuf[a * 512 + out * 5 + kd]);
    acc += expf(-norm);
  }
  ob[bb * 100 + out] = acc;
}

// ---------------- final linear + sigmoid: one block per batch row -----------
__global__ void k_final(const float* __restrict__ feat32, const float* __restrict__ ob,
                        const float* __restrict__ lw, const float* __restrict__ lb,
                        float* __restrict__ out) {
  __shared__ float red[256];
  int b = blockIdx.x;
  float acc = 0.f;
  const float* f = feat32 + (size_t)b * FLAT;
  for (int i = threadIdx.x; i < FLAT; i += 256) acc += f[i] * lw[i];
  for (int i = threadIdx.x; i < 100; i += 256)  acc += ob[b * 100 + i] * lw[FLAT + i];
  red[threadIdx.x] = acc;
  __syncthreads();
  for (int off = 128; off > 0; off >>= 1) {
    if (threadIdx.x < off) red[threadIdx.x] += red[threadIdx.x + off];
    __syncthreads();
  }
  if (threadIdx.x == 0) {
    float logit = red[0] + lb[0];
    out[b] = 1.f / (1.f + expf(-logit));
  }
}

extern "C" void kernel_launch(void* const* d_in, const int* in_sizes, int n_in,
                              void* d_out, int out_size, void* d_ws, size_t ws_size,
                              hipStream_t stream) {
  const float* x       = (const float*)d_in[0];
  const float* conv1_w = (const float*)d_in[1];
  const float* conv1_b = (const float*)d_in[2];
  const float* conv2_w = (const float*)d_in[3];
  const float* conv2_b = (const float*)d_in[4];
  const float* gamma   = (const float*)d_in[5];
  const float* beta    = (const float*)d_in[6];
  const float* T       = (const float*)d_in[7];
  const float* lin_w   = (const float*)d_in[8];
  const float* lin_b   = (const float*)d_in[9];
  float* out = (float*)d_out;

  // workspace carving (256B aligned)
  char* base = (char*)d_ws;
  size_t off = 0;
  auto carve = [&](size_t bytes) -> void* {
    void* p = base + off;
    off += (bytes + 255) & ~(size_t)255;
    return p;
  };
  _Float16* h1f16  = (_Float16*)carve((size_t)BATCH * 64 * 4096 * 2);   // 33.5 MB
  _Float16* wAh    = (_Float16*)carve((size_t)128 * 1024 * 2);          // 0.25 MB
  float*    h2     = (float*)   carve((size_t)BATCH * FLAT * 4);        // 33.5 MB
  _Float16* feat16 = (_Float16*)carve((size_t)BATCH * FLAT * 2);        // 16.8 MB
  _Float16* Tt     = (_Float16*)carve((size_t)512 * FLAT * 2);          // 134 MB
  float*    Mbuf   = (float*)   carve((size_t)BATCH * 512 * 4);
  float*    ob     = (float*)   carve((size_t)BATCH * 100 * 4);
  float*    mu     = (float*)   carve(128 * 4);
  float*    rstd   = (float*)   carve(128 * 4);
  (void)ws_size; (void)n_in; (void)in_sizes; (void)out_size;

  // conv1 (leaky, f16 out)
  k_conv1<<<65536, 256, 0, stream>>>(x, conv1_w, conv1_b, h1f16);
  // conv2 weight cvt + implicit-GEMM WMMA conv2
  k_cvtW<<<512, 256, 0, stream>>>(conv2_w, wAh);
  k_conv2<<<4096, 256, 0, stream>>>(h1f16, wAh, conv2_b, h2);
  // batchnorm (batch stats) + leaky, produce f32 feat (in place) and f16 feat
  k_bnstats<<<128, 256, 0, stream>>>(h2, mu, rstd);
  k_bnapply<<<32768, 256, 0, stream>>>(h2, feat16, mu, rstd, gamma, beta);
  // minibatch discrimination GEMM: transpose-cvt T, zero M, K-split WMMA
  k_cvtT<<<dim3(4096, 16), dim3(32, 8), 0, stream>>>(T, Tt);
  k_zero<<<128, 256, 0, stream>>>(Mbuf, BATCH * 512);
  k_mbd_gemm<<<dim3(8, 64), 32, 0, stream>>>(feat16, Tt, Mbuf);
  // pairwise L1 / exp / sum
  k_ob<<<100, 64, 0, stream>>>(Mbuf, ob);
  // final linear + sigmoid
  k_final<<<64, 256, 0, stream>>>(h2, ob, lin_w, lin_b, out);
}
